// Net_22402549416295
// MI455X (gfx1250) — matrix operand order repaired and checked by reading, not compiled, not run on hardware
//
#include <hip/hip_runtime.h>

typedef __bf16 bf16_t;
typedef __attribute__((ext_vector_type(16))) bf16_t v16bf;
typedef __attribute__((ext_vector_type(4)))  bf16_t v4bf;
typedef __attribute__((ext_vector_type(8)))  float  v8f;
typedef __attribute__((ext_vector_type(4)))  float  v4f;

#define H160 160

__device__ __forceinline__ float silu_f(float x){ return x / (1.0f + __expf(-x)); }

__device__ __forceinline__ v8f wmma_bf16(v16bf a, v16bf b, v8f c){
  return __builtin_amdgcn_wmma_f32_16x16x32_bf16(false, a, false, b, (short)0, c, false, false);
}

// B-fragment load from pre-swizzled weights: frag = 512 bf16 (1KB), lane gets 32B contiguous
__device__ __forceinline__ v16bf ldB(const bf16_t* __restrict__ wsw, int frag, int lane){
  return ((const v16bf*)(wsw + (size_t)frag * 512))[lane];
}

// A-fragment from bf16 LDS row-major tile (16x32 tile at rows mbase.., cols kb..)
// Lane L: m = mbase + (L&15); half = L>>4 ; e<8 -> K = kb+8*half+e ; e>=8 -> K = kb+16+8*half+(e-8)
__device__ __forceinline__ v16bf a_frag_bf16(const bf16_t* As, int stride, int mbase, int kb, int lane){
  int m = mbase + (lane & 15);
  int h8 = (lane >> 4) * 8;
  const bf16_t* row = As + m * stride;
  v16bf a;
#pragma unroll
  for (int e = 0; e < 8; ++e) a[e] = row[kb + h8 + e];
#pragma unroll
  for (int e = 0; e < 8; ++e) a[8 + e] = row[kb + 16 + h8 + e];
  return a;
}

// A-fragment = bf16 LDS + fp32 LDS (adf + delta) for GEMM3
__device__ __forceinline__ v16bf a_frag_sum(const bf16_t* adf, const float* delta,
                                            int mbase, int kb, int lane){
  int m = mbase + (lane & 15);
  int h8 = (lane >> 4) * 8;
  const bf16_t* ar = adf + m * 168;
  const float*  dr = delta + m * 164;
  v16bf a;
#pragma unroll
  for (int e = 0; e < 8; ++e){ int k = kb + h8 + e;      a[e]     = (bf16_t)((float)ar[k] + dr[k]); }
#pragma unroll
  for (int e = 0; e < 8; ++e){ int k = kb + 16 + h8 + e; a[8 + e] = (bf16_t)((float)ar[k] + dr[k]); }
  return a;
}

// ---------------- weight swizzle: fp32 [K x N] -> bf16 fragments --------------------
// frag f = nt*(K/32) + kt ; lane L: n = nt*16+(L&15); element e: e<8 -> K = kt*32 + (L>>4)*8 + e ; e>=8 -> K = kt*32 + 16 + (L>>4)*8 + (e-8)
__global__ void swz_weight_kernel(const float* __restrict__ W, bf16_t* __restrict__ out, int K, int N){
  int f = blockIdx.x;
  int lane = threadIdx.x;
  int ktc = K >> 5;
  int nt = f / ktc, kt = f - nt * ktc;
  int n = nt * 16 + (lane & 15);
  int kbase = kt * 32 + (lane >> 4) * 8;
  bf16_t* o = out + (size_t)f * 512 + lane * 16;
#pragma unroll
  for (int e = 0; e < 16; ++e) o[e] = (bf16_t)W[(size_t)(kbase + e + (e >= 8 ? 8 : 0)) * N + n];
}

// ---------------- generic WMMA GEMM: C[M,N] = act(A[M,K] @ W + bias) ----------------
// grid.x = M/32 ; grid.y = N/160 ; block = 320 (10 waves, one 16-col tile each)
// A tile staged in LDS as bf16 via 16B vector loads; fragments are pure ds_load_b128.
__global__ __launch_bounds__(320) void gemm_bf16_kernel(
    const float* __restrict__ A, const bf16_t* __restrict__ Wsw,
    const float* __restrict__ bias, float* __restrict__ C,
    int M, int N, int K, int act){
  __shared__ bf16_t As[32 * 168];
  int tid = threadIdx.x;
  int wave = tid >> 5, lane = tid & 31;
  int mg = blockIdx.x * 32;
  int ntile = blockIdx.y * 10 + wave;
  int nb = ntile * 16;
  int ktc = K >> 5;
  const bf16_t* wbase = Wsw + (size_t)ntile * ktc * 512;
  v8f acc0 = {}; v8f acc1 = {};
  for (int k0 = 0; k0 < K; k0 += 160){
    __syncthreads();
    for (int idx = tid; idx < 32 * 40; idx += 320){
      int r = idx / 40, c = (idx - r * 40) * 4;
      v4f x = *(const v4f*)(A + (size_t)(mg + r) * K + k0 + c);
      v4bf y;
#pragma unroll
      for (int e = 0; e < 4; ++e) y[e] = (bf16_t)x[e];
      *(v4bf*)(As + r * 168 + c) = y;
    }
    __syncthreads();
    int kg0 = k0 >> 5;
#pragma unroll
    for (int kt = 0; kt < 5; ++kt){
      if (kg0 + kt + 1 < ktc)
        __builtin_prefetch(wbase + (size_t)(kg0 + kt + 1) * 512 + lane * 16, 0, 0);
      v16bf bfrag = ldB(wbase, kg0 + kt, lane);
      acc0 = wmma_bf16(a_frag_bf16(As, 168, 0,  kt * 32, lane), bfrag, acc0);
      acc1 = wmma_bf16(a_frag_bf16(As, 168, 16, kt * 32, lane), bfrag, acc1);
    }
  }
  int n = lane & 15, half = lane >> 4;
  float bv = bias ? bias[nb + n] : 0.0f;
#pragma unroll
  for (int r = 0; r < 8; ++r){
    float x0 = acc0[r] + bv;
    float x1 = acc1[r] + bv;
    if (act){ x0 = silu_f(x0); x1 = silu_f(x1); }
    C[(size_t)(mg + half * 8 + r) * N + nb + n] = x0;
    C[(size_t)(mg + 16 + half * 8 + r) * N + nb + n] = x1;
  }
}

// ---------------- fused point-transformer-conv edge kernel --------------------------
// One block (10 waves) per destination point; KE edges = M dimension.
template<int KE>
__global__ __launch_bounds__(320) void ptc_edge_kernel(
    const float* __restrict__ pos_dst, const float* __restrict__ pos_src,
    const float* __restrict__ a_dst,   const float* __restrict__ a_src,
    const float* __restrict__ vsrc,    const int* __restrict__ nbr,
    const bf16_t* __restrict__ Wp1, const float* __restrict__ bp1,
    const bf16_t* __restrict__ Wp2, const float* __restrict__ bp2,
    const bf16_t* __restrict__ Wa,  const float* __restrict__ ba,
    float* __restrict__ outp, int nsrcPer, int ndstPer){
  extern __shared__ char smem_raw[];
  bf16_t* de   = (bf16_t*)smem_raw;                 // KE x 168 bf16  (pos_dst - pos_j)
  bf16_t* h1   = de  + KE * 168;                    // KE x 168 bf16
  bf16_t* adf  = h1  + KE * 168;                    // KE x 168 bf16  (a_dst - a_src_j)
  float*  delta = (float*)(smem_raw + (size_t)KE * 1008); // KE x 164 f32
  float*  vj    = delta + KE * 164;                 // KE x 164 f32

  int dst = blockIdx.x;
  int b = dst / ndstPer;
  int tid = threadIdx.x, wave = tid >> 5, lane = tid & 31;
  int nb = wave * 16, n = lane & 15, half = lane >> 4;

  const float* pd = pos_dst + (size_t)dst * H160;
  const float* ad = a_dst   + (size_t)dst * H160;
  // phase 0: gather edge inputs, 16B vectors
  for (int idx = tid; idx < KE * 40; idx += 320){
    int k = idx / 40, h = (idx - k * 40) * 4;
    int j = nbr[(size_t)dst * KE + k];
    size_t sr = (size_t)(b * nsrcPer + j) * H160 + h;
    v4f pdv = *(const v4f*)(pd + h);
    v4f adv = *(const v4f*)(ad + h);
    v4f psv = *(const v4f*)(pos_src + sr);
    v4f asv = *(const v4f*)(a_src + sr);
    v4f vv  = *(const v4f*)(vsrc + sr);
    v4bf dv, av;
#pragma unroll
    for (int e = 0; e < 4; ++e){
      dv[e] = (bf16_t)(pdv[e] - psv[e]);
      av[e] = (bf16_t)(adv[e] - asv[e]);
    }
    *(v4bf*)(de  + k * 168 + h) = dv;
    *(v4bf*)(adf + k * 168 + h) = av;
    *(v4f*)(vj + k * 164 + h) = vv;
  }
  __syncthreads();

  // GEMM1: h1 = silu(de @ Wp1 + bp1)
  {
    v8f acc0 = {}; v8f acc1 = {};
#pragma unroll
    for (int kt = 0; kt < 5; ++kt){
      v16bf bfrag = ldB(Wp1, wave * 5 + kt, lane);
      acc0 = wmma_bf16(a_frag_bf16(de, 168, 0, kt * 32, lane), bfrag, acc0);
      if (KE == 32) acc1 = wmma_bf16(a_frag_bf16(de, 168, 16, kt * 32, lane), bfrag, acc1);
    }
    float bv = bp1[nb + n];
#pragma unroll
    for (int r = 0; r < 8; ++r){
      h1[(half * 8 + r) * 168 + nb + n] = (bf16_t)silu_f(acc0[r] + bv);
      if (KE == 32) h1[(16 + half * 8 + r) * 168 + nb + n] = (bf16_t)silu_f(acc1[r] + bv);
    }
  }
  __syncthreads();

  // GEMM2: delta = h1 @ Wp2 + bp2
  {
    v8f acc0 = {}; v8f acc1 = {};
#pragma unroll
    for (int kt = 0; kt < 5; ++kt){
      v16bf bfrag = ldB(Wp2, wave * 5 + kt, lane);
      acc0 = wmma_bf16(a_frag_bf16(h1, 168, 0, kt * 32, lane), bfrag, acc0);
      if (KE == 32) acc1 = wmma_bf16(a_frag_bf16(h1, 168, 16, kt * 32, lane), bfrag, acc1);
    }
    float bv = bp2[nb + n];
#pragma unroll
    for (int r = 0; r < 8; ++r){
      delta[(half * 8 + r) * 164 + nb + n] = acc0[r] + bv;
      if (KE == 32) delta[(16 + half * 8 + r) * 164 + nb + n] = acc1[r] + bv;
    }
  }
  __syncthreads();

  // GEMM3: alpha = (adf + delta) @ Wa + ba ; softmax over KE edges per channel; out
  {
    v8f acc0 = {}; v8f acc1 = {};
#pragma unroll
    for (int kt = 0; kt < 5; ++kt){
      v16bf bfrag = ldB(Wa, wave * 5 + kt, lane);
      acc0 = wmma_bf16(a_frag_sum(adf, delta, 0, kt * 32, lane), bfrag, acc0);
      if (KE == 32) acc1 = wmma_bf16(a_frag_sum(adf, delta, 16, kt * 32, lane), bfrag, acc1);
    }
    const int NL = (KE == 32) ? 16 : 8;
    float bv = ba[nb + n];
    float al[16];
#pragma unroll
    for (int r = 0; r < 8; ++r) al[r] = acc0[r] + bv;
    if (KE == 32){
#pragma unroll
      for (int r = 0; r < 8; ++r) al[8 + r] = acc1[r] + bv;
    }
    float mx = al[0];
#pragma unroll
    for (int i = 1; i < 16; ++i) if (i < NL) mx = fmaxf(mx, al[i]);
    mx = fmaxf(mx, __shfl_xor(mx, 16, 32));
    float s = 0.0f;
#pragma unroll
    for (int i = 0; i < 16; ++i) if (i < NL){ al[i] = __expf(al[i] - mx); s += al[i]; }
    s += __shfl_xor(s, 16, 32);
    float inv = 1.0f / s;
    float o = 0.0f;
#pragma unroll
    for (int r = 0; r < 8; ++r){
      int m = half * 8 + r;
      o += al[r] * (vj[m * 164 + nb + n] + delta[m * 164 + nb + n]);
    }
    if (KE == 32){
#pragma unroll
      for (int r = 0; r < 8; ++r){
        int m = 16 + half * 8 + r;
        o += al[8 + r] * (vj[m * 164 + nb + n] + delta[m * 164 + nb + n]);
      }
    }
    o *= inv;
    o += __shfl_xor(o, 16, 32);
    if (half == 0) outp[(size_t)dst * H160 + nb + n] = o;
  }
}

// ---------------- kNN: one block per dst point; blockDim = nsrcPer ------------------
__global__ void knn_kernel(const float* __restrict__ pos_dst, const float* __restrict__ pos_src,
                           int ndstPer, int nsrcPer, int k, int* __restrict__ outIdx){
  __shared__ __align__(16) float pd[H160];
  __shared__ float dist[256];
  __shared__ float wd[256];
  __shared__ int   wi[256];
  int dstId = blockIdx.x;
  int b = dstId / ndstPer;
  int t = threadIdx.x;
  int ns = blockDim.x;
  for (int h = t; h < H160; h += ns) pd[h] = pos_dst[(size_t)dstId * H160 + h];
  __syncthreads();
  const float* ps = pos_src + (size_t)(b * nsrcPer + t) * H160;
  float d = 0.0f;
  for (int q = 0; q < 40; ++q){
    v4f a4 = *(const v4f*)(pd + q * 4);
    v4f b4 = *(const v4f*)(ps + q * 4);
#pragma unroll
    for (int e = 0; e < 4; ++e){ float df = a4[e] - b4[e]; d += df * df; }
  }
  dist[t] = d;
  __syncthreads();
  for (int it = 0; it < k; ++it){
    wd[t] = dist[t]; wi[t] = t;
    __syncthreads();
    for (int s = ns >> 1; s > 0; s >>= 1){
      if (t < s){
        float d2 = wd[t + s]; int i2 = wi[t + s];
        if (d2 < wd[t] || (d2 == wd[t] && i2 < wi[t])){ wd[t] = d2; wi[t] = i2; }
      }
      __syncthreads();
    }
    if (t == 0){ outIdx[(size_t)dstId * k + it] = wi[0]; dist[wi[0]] = 3.0e38f; }
    __syncthreads();
  }
}

// ---------------- particle encoders: init (3-layer) + orig (1-layer), mask-selected -
__global__ void enc_kernel(const float* __restrict__ x_pfc,
    const float* chW1, const float* chB1, const float* chW2, const float* chB2,
    const float* chW3, const float* chB3,
    const float* neW1, const float* neB1, const float* neW2, const float* neB2,
    const float* neW3, const float* neB3,
    const float* ch2W, const float* ch2B, const float* ne2W, const float* ne2B,
    float* __restrict__ initOut, float* __restrict__ origOut){
  __shared__ float x[12];
  __shared__ float hh1[80];
  __shared__ float hh2[H160];
  int row = blockIdx.x;
  int t = threadIdx.x; // 160
  if (t < 12) x[t] = x_pfc[(size_t)row * 12 + t];
  __syncthreads();
  bool ch = (x[10] != 0.0f);
  int K = ch ? 12 : 11;
  const float *W1 = ch ? chW1 : neW1, *B1 = ch ? chB1 : neB1;
  const float *W2 = ch ? chW2 : neW2, *B2 = ch ? chB2 : neB2;
  const float *W3 = ch ? chW3 : neW3, *B3 = ch ? chB3 : neB3;
  if (t < 80){
    float s = B1[t];
    for (int k = 0; k < K; ++k) s += x[k] * W1[k * 80 + t];
    hh1[t] = silu_f(s);
  }
  __syncthreads();
  {
    float s = B2[t];
    for (int k = 0; k < 80; ++k) s += hh1[k] * W2[k * H160 + t];
    hh2[t] = silu_f(s);
  }
  __syncthreads();
  {
    float s = B3[t];
    for (int k = 0; k < H160; ++k) s += hh2[k] * W3[k * H160 + t];
    initOut[(size_t)row * H160 + t] = s;
  }
  {
    const float* W = ch ? ch2W : ne2W;
    const float* Bb = ch ? ch2B : ne2B;
    float s = Bb[t];
    for (int k = 0; k < K; ++k) s += x[k] * W[k * H160 + t];
    origOut[(size_t)row * H160 + t] = s;
  }
}

// ---------------- vtx_enc1: [5 -> 40 -> 80 -> 160 -> 160] ---------------------------
__global__ void vtx1_kernel(const float* __restrict__ x_vtx,
    const float* W1, const float* B1, const float* W2, const float* B2,
    const float* W3, const float* B3, const float* W4, const float* B4,
    float* __restrict__ out){
  __shared__ float x[5];
  __shared__ float a[H160];
  __shared__ float bb[H160];
  int row = blockIdx.x;
  int t = threadIdx.x; // 160
  if (t < 5) x[t] = x_vtx[(size_t)row * 5 + t];
  __syncthreads();
  if (t < 40){ float s = B1[t]; for (int k = 0; k < 5;  ++k) s += x[k]  * W1[k * 40 + t];  a[t]  = silu_f(s); }
  __syncthreads();
  if (t < 80){ float s = B2[t]; for (int k = 0; k < 40; ++k) s += a[k]  * W2[k * 80 + t];  bb[t] = silu_f(s); }
  __syncthreads();
  { float s = B3[t]; for (int k = 0; k < 80;  ++k) s += bb[k] * W3[k * H160 + t]; a[t] = silu_f(s); }
  __syncthreads();
  { float s = B4[t]; for (int k = 0; k < H160; ++k) s += a[k]  * W4[k * H160 + t]; out[(size_t)row * H160 + t] = s; }
}

// ---------------- compaction of charged indices (ordered, like jnp.nonzero) ---------
__global__ void cidx_kernel(const float* __restrict__ x_pfc, int* __restrict__ cidx){
  __shared__ int flags[256];
  __shared__ int base;
  int t = threadIdx.x;
  if (t == 0) base = 0;
  __syncthreads();
  for (int chunk = 0; chunk < 8192; chunk += 256){
    int i = chunk + t;
    int f = (x_pfc[(size_t)i * 12 + 10] != 0.0f) ? 1 : 0;
    flags[t] = f;
    __syncthreads();
    for (int off = 1; off < 256; off <<= 1){
      int add = (t >= off) ? flags[t - off] : 0;
      __syncthreads();
      flags[t] += add;
      __syncthreads();
    }
    if (f){
      int pos = base + flags[t] - 1;
      if (pos < 4096) cidx[pos] = i;
    }
    __syncthreads();
    if (t == 255) base += flags[255];
    __syncthreads();
  }
}

__global__ void gather_rows_kernel(const float* __restrict__ src, const int* __restrict__ cidx,
                                   float* __restrict__ dst){
  int i = blockIdx.x;
  int r = cidx[i];
  const v4f* s = (const v4f*)(src + (size_t)r * H160);
  v4f* d = (v4f*)(dst + (size_t)i * H160);
  for (int q = threadIdx.x; q < 40; q += blockDim.x) d[q] = s[q];
}

// ---------------- final score: s = dot(pfc_final[i], vtx_final[8*(i/256)]) ----------
__global__ void score_kernel(const float* __restrict__ pfcfin, const float* __restrict__ vtxfin,
                             float* __restrict__ scores){
  __shared__ float red[H160];
  int i = blockIdx.x, t = threadIdx.x;
  const float* tg = vtxfin + (size_t)(i >> 8) * 8 * H160;
  red[t] = pfcfin[(size_t)i * H160 + t] * tg[t];
  __syncthreads();
  if (t < 80) red[t] += red[t + 80]; __syncthreads();
  if (t < 40) red[t] += red[t + 40]; __syncthreads();
  if (t < 20) red[t] += red[t + 20]; __syncthreads();
  if (t < 10) red[t] += red[t + 10]; __syncthreads();
  if (t < 5)  red[t] += red[t + 5];  __syncthreads();
  if (t == 0){
    float s = red[0] + red[1] + red[2] + red[3] + red[4];
    scores[2 * (size_t)i] = s;
    scores[2 * (size_t)i + 1] = -s;
  }
}

// =====================================================================================
extern "C" void kernel_launch(void* const* d_in, const int* in_sizes, int n_in,
                              void* d_out, int out_size, void* d_ws, size_t ws_size,
                              hipStream_t stream){
  (void)in_sizes; (void)n_in; (void)out_size; (void)ws_size;
  auto IN = [&](int i)->const float*{ return (const float*)d_in[i]; };

  // ---- workspace bump allocator ----
  char* wsbase = (char*)d_ws;
  size_t off = 0;
  auto alloc = [&](size_t bytes)->void*{
    void* p = wsbase + off;
    off += (bytes + 255) & ~(size_t)255;
    return p;
  };

  // ---- swizzle weights to bf16 fragments ----
  auto swz = [&](const float* W, int K, int N)->const bf16_t*{
    bf16_t* dst = (bf16_t*)alloc((size_t)K * N * sizeof(bf16_t));
    int frags = (N / 16) * (K / 32);
    swz_weight_kernel<<<frags, 32, 0, stream>>>(W, dst, K, N);
    return dst;
  };

  struct ConvSw { const bf16_t *Wlin, *Wsrc, *Wdst, *Wp1, *Wp2, *Wa; const float *bp1, *bp2, *ba; };
  auto mkconv = [&](int base)->ConvSw{
    ConvSw c;
    c.Wlin = swz(IN(base + 0), 160, 160);
    c.Wsrc = swz(IN(base + 1), 160, 160);
    c.Wdst = swz(IN(base + 2), 160, 160);
    c.Wp1  = swz(IN(base + 3), 160, 160); c.bp1 = IN(base + 4);
    c.Wp2  = swz(IN(base + 5), 160, 160); c.bp2 = IN(base + 6);
    c.Wa   = swz(IN(base + 7), 160, 160); c.ba  = IN(base + 8);
    return c;
  };
  struct MlpSw { const bf16_t* W[3]; const float* Bb[3]; };
  auto mkmlp = [&](int base)->MlpSw{
    MlpSw m;
    m.W[0] = swz(IN(base + 0), 160, 320); m.Bb[0] = IN(base + 1);
    m.W[1] = swz(IN(base + 2), 320, 640); m.Bb[1] = IN(base + 3);
    m.W[2] = swz(IN(base + 4), 640, 160); m.Bb[2] = IN(base + 5);
    return m;
  };

  // input indices (pytree leaf order)
  const float* x_pfc = IN(0);
  const float* x_vtx = IN(1);
  ConvSw c1a = mkconv(38), c1b = mkconv(47), c2a = mkconv(56), c2b = mkconv(65);
  MlpSw  mvtx = mkmlp(26), mpfc = mkmlp(32);

  // ---- buffers ----
  const size_t R160 = (size_t)8192 * 160 * 4;
  float* buf_init   = (float*)alloc(R160);
  float* buf_e1     = (float*)alloc(R160);
  float* buf_orig   = (float*)alloc(R160);
  float* buf_as     = (float*)alloc(R160);
  float* buf_ad     = (float*)alloc(R160);
  float* buf_v      = (float*)alloc(R160);
  float* buf_f2a    = (float*)alloc(R160);
  float* buf_chA    = (float*)alloc((size_t)4096 * 160 * 4);
  float* buf_chB    = (float*)alloc((size_t)4096 * 160 * 4);
  float* buf_h1     = (float*)alloc((size_t)8192 * 320 * 4);
  float* buf_h2     = (float*)alloc((size_t)8192 * 640 * 4);
  float* buf_pfcfin = (float*)alloc(R160);
  int*   nbr        = (int*)alloc((size_t)8192 * 32 * 4);
  int*   cidx       = (int*)alloc((size_t)4096 * 4);

  float* out_scores = (float*)d_out;
  float* out_euc    = out_scores + 16384;
  float* out_vtx    = out_euc + (size_t)8192 * 160;

  auto gemm = [&](const float* A, const bf16_t* Wsw, const float* bias, float* C,
                  int M, int N, int K, int act){
    dim3 grid(M / 32, N / 160);
    gemm_bf16_kernel<<<grid, 320, 0, stream>>>(A, Wsw, bias, C, M, N, K, act);
  };
  const size_t shm32 = (size_t)32 * 2320;
  const size_t shm16 = (size_t)16 * 2320;

  // 1) vertex encoder 1 -> out_vtx (also an output)
  vtx1_kernel<<<256, 160, 0, stream>>>(x_vtx, IN(2), IN(3), IN(4), IN(5), IN(6), IN(7), IN(8), IN(9), out_vtx);

  // 2) particle encoders -> init, orig
  enc_kernel<<<8192, 160, 0, stream>>>(x_pfc,
      IN(10), IN(11), IN(12), IN(13), IN(14), IN(15),
      IN(16), IN(17), IN(18), IN(19), IN(20), IN(21),
      IN(22), IN(23), IN(24), IN(25),
      buf_init, buf_orig);

  // 3) conv1a
  knn_kernel<<<8192, 256, 0, stream>>>(buf_init, buf_init, 256, 256, 32, nbr);
  gemm(buf_init, c1a.Wsrc, nullptr, buf_as, 8192, 160, 160, 0);
  gemm(buf_init, c1a.Wdst, nullptr, buf_ad, 8192, 160, 160, 0);
  gemm(buf_init, c1a.Wlin, nullptr, buf_v,  8192, 160, 160, 0);
  ptc_edge_kernel<32><<<8192, 320, shm32, stream>>>(buf_init, buf_init, buf_ad, buf_as, buf_v,
      nbr, c1a.Wp1, c1a.bp1, c1a.Wp2, c1a.bp2, c1a.Wa, c1a.ba, buf_e1, 256, 256);

  // 4) conv1b -> euc (output slice)
  knn_kernel<<<8192, 256, 0, stream>>>(buf_e1, buf_e1, 256, 256, 32, nbr);
  gemm(buf_e1, c1b.Wsrc, nullptr, buf_as, 8192, 160, 160, 0);
  gemm(buf_e1, c1b.Wdst, nullptr, buf_ad, 8192, 160, 160, 0);
  gemm(buf_e1, c1b.Wlin, nullptr, buf_v,  8192, 160, 160, 0);
  ptc_edge_kernel<32><<<8192, 320, shm32, stream>>>(buf_e1, buf_e1, buf_ad, buf_as, buf_v,
      nbr, c1b.Wp1, c1b.bp1, c1b.Wp2, c1b.bp2, c1b.Wa, c1b.ba, out_euc, 256, 256);

  // 5) charged index compaction + gathers
  cidx_kernel<<<1, 256, 0, stream>>>(x_pfc, cidx);
  gather_rows_kernel<<<4096, 64, 0, stream>>>(buf_orig, cidx, buf_chA); // ch_orig
  gather_rows_kernel<<<4096, 64, 0, stream>>>(out_euc,  cidx, buf_chB); // ch_pos

  // 6) conv2a: src = charged (128/batch), dst = all (256/batch)
  knn_kernel<<<8192, 128, 0, stream>>>(out_euc, buf_chB, 256, 128, 16, nbr);
  gemm(buf_chA, c2a.Wsrc, nullptr, buf_as, 4096, 160, 160, 0);
  gemm(buf_orig, c2a.Wdst, nullptr, buf_ad, 8192, 160, 160, 0);
  gemm(buf_chA, c2a.Wlin, nullptr, buf_v,  4096, 160, 160, 0);
  ptc_edge_kernel<16><<<8192, 320, shm16, stream>>>(out_euc, buf_chB, buf_ad, buf_as, buf_v,
      nbr, c2a.Wp1, c2a.bp1, c2a.Wp2, c2a.bp2, c2a.Wa, c2a.ba, buf_f2a, 128, 256);

  // 7) conv2b
  gather_rows_kernel<<<4096, 64, 0, stream>>>(buf_f2a, cidx, buf_chA); // ch_f2
  knn_kernel<<<8192, 128, 0, stream>>>(buf_f2a, buf_chA, 256, 128, 16, nbr);
  gemm(buf_chA, c2b.Wsrc, nullptr, buf_as, 4096, 160, 160, 0);
  gemm(buf_f2a, c2b.Wdst, nullptr, buf_ad, 8192, 160, 160, 0);
  gemm(buf_chA, c2b.Wlin, nullptr, buf_v,  4096, 160, 160, 0);
  ptc_edge_kernel<16><<<8192, 320, shm16, stream>>>(buf_f2a, buf_chA, buf_ad, buf_as, buf_v,
      nbr, c2b.Wp1, c2b.bp1, c2b.Wp2, c2b.bp2, c2b.Wa, c2b.ba, buf_init, 128, 256); // f2 final

  // 8) pfc_enc2 MLP (WMMA): 160 -> 320 -> 640 -> 160
  gemm(buf_init, mpfc.W[0], mpfc.Bb[0], buf_h1, 8192, 320, 160, 1);
  gemm(buf_h1,   mpfc.W[1], mpfc.Bb[1], buf_h2, 8192, 640, 320, 1);
  gemm(buf_h2,   mpfc.W[2], mpfc.Bb[2], buf_pfcfin, 8192, 160, 640, 0);

  // 9) vtx_enc2 MLP (WMMA) on x_vtx_euc (reuse buffers)
  gemm(out_vtx, mvtx.W[0], mvtx.Bb[0], buf_as, 256, 320, 160, 1);
  gemm(buf_as,  mvtx.W[1], mvtx.Bb[1], buf_ad, 256, 640, 320, 1);
  gemm(buf_ad,  mvtx.W[2], mvtx.Bb[2], buf_v,  256, 160, 640, 0);

  // 10) scores
  score_kernel<<<8192, 160, 0, stream>>>(buf_pfcfin, buf_v, out_scores);
}
